// Attention_46626164965681
// MI455X (gfx1250) — matrix operand order repaired and checked by reading, not compiled, not run on hardware
//
#include <hip/hip_runtime.h>
#include <hip/hip_bf16.h>

typedef __attribute__((ext_vector_type(16))) _Float16 v16h;
typedef __attribute__((ext_vector_type(8)))  _Float16 v8h;
typedef __attribute__((ext_vector_type(8)))  float    v8f;

#define HIDDEN  1024
#define NB      4
#define SEQ     1024
#define NHEADS  16
#define DEPTH   64

static __device__ __forceinline__ v16h hcat(v8h lo, v8h hi) {
  return __builtin_shufflevector(lo, hi, 0,1,2,3,4,5,6,7,8,9,10,11,12,13,14,15);
}
static __device__ __forceinline__ v8f wmma_f16(v16h a, v16h b, v8f c) {
  return __builtin_amdgcn_wmma_f32_16x16x32_f16(false, a, false, b, (short)0, c, false, false);
}

// gfx1250 async global->LDS copy (16B per lane), tracked by ASYNCcnt.
static __device__ __forceinline__ void async_copy_b128(const void* gsrc,
                                                       void* lds_dst) {
  unsigned int loff = (unsigned int)(uintptr_t)lds_dst;  // addr[31:0] = LDS offset
  unsigned long long ga = (unsigned long long)(uintptr_t)gsrc;
  asm volatile("global_load_async_to_lds_b128 %0, %1, off"
               :: "v"(loff), "v"(ga) : "memory");
}
static __device__ __forceinline__ void wait_async() {
  asm volatile("s_wait_asynccnt 0x0" ::: "memory");
}

// ---------------------------------------------------------------------------
// Kernel 1: convert fp32 weights [k][n] -> fp16 transposed WT[n][k]
// grid (32,32,4), block (32,8)
// ---------------------------------------------------------------------------
__global__ void __launch_bounds__(256) transpose_w_kernel(
    const float* __restrict__ Wq, const float* __restrict__ Wk,
    const float* __restrict__ Wv, const float* __restrict__ Wo,
    _Float16* __restrict__ wt) {
  __shared__ float tile[32][33];
  const float* W = (blockIdx.z == 0) ? Wq : (blockIdx.z == 1) ? Wk
                 : (blockIdx.z == 2) ? Wv : Wo;
  _Float16* out = wt + (size_t)blockIdx.z * (HIDDEN * HIDDEN);
  int tx = threadIdx.x, ty = threadIdx.y;
  int nb = blockIdx.x * 32, kb = blockIdx.y * 32;
#pragma unroll
  for (int i = 0; i < 4; i++)
    tile[ty + i * 8][tx] = W[(size_t)(kb + ty + i * 8) * HIDDEN + nb + tx];
  __syncthreads();
#pragma unroll
  for (int i = 0; i < 4; i++)
    out[(size_t)(nb + ty + i * 8) * HIDDEN + kb + tx] =
        (_Float16)tile[tx][ty + i * 8];
}

// ---------------------------------------------------------------------------
// Kernel 2: fused QKV projection GEMM.  M=4096, N=1024, K=1024.
// grid (N/128=8, M/128=32, 3), block 256 = 8 waves (2M x 4N, 64x32 per wave).
// Double-buffered LDS; B panel via async-to-LDS; A panel fp32->fp16 staged.
// Q,K written [B,h,S,d] fp16 (Q pre-scaled by 1/8); V written [B,h,d,S] fp16.
// ---------------------------------------------------------------------------
__global__ void __launch_bounds__(256) qkv_proj_kernel(
    const float* __restrict__ x, const float* __restrict__ y,
    const _Float16* __restrict__ wt,
    _Float16* __restrict__ qout, _Float16* __restrict__ kout,
    _Float16* __restrict__ vtout) {
  __shared__ _Float16 As[2][128 * 32];
  __shared__ _Float16 Bs[2][128 * 32];
  const int w = blockIdx.z;
  const float* src = (w == 0) ? x : y;
  const _Float16* wtw = wt + (size_t)w * (HIDDEN * HIDDEN);
  const int mbase = blockIdx.y * 128, nbase = blockIdx.x * 128;
  const int tid = threadIdx.x, wave = tid >> 5, lane = tid & 31;
  const int lh = lane & 15, hig = lane >> 4;
  const int waveM = wave >> 2, waveN = wave & 3;
  const int fr = tid >> 1, fseg = (tid & 1) * 16;  // fill coords (16 elems/thread)

  v8f z = {0.f,0.f,0.f,0.f,0.f,0.f,0.f,0.f};
  v8f C[4][2];
#pragma unroll
  for (int mt = 0; mt < 4; mt++) { C[mt][0] = z; C[mt][1] = z; }

  // ---- prologue: fill buffer 0
  {
    const float4* s4 = (const float4*)(src + (size_t)(mbase + fr) * HIDDEN + fseg);
    float4 f0 = s4[0], f1 = s4[1], f2 = s4[2], f3 = s4[3];
    const _Float16* g = wtw + (size_t)(nbase + fr) * HIDDEN + fseg;
    _Float16* l = &Bs[0][fr * 32 + fseg];
    async_copy_b128(g, l);
    async_copy_b128(g + 8, l + 8);
    v8h h0, h1;
    h0[0]=(_Float16)f0.x; h0[1]=(_Float16)f0.y; h0[2]=(_Float16)f0.z; h0[3]=(_Float16)f0.w;
    h0[4]=(_Float16)f1.x; h0[5]=(_Float16)f1.y; h0[6]=(_Float16)f1.z; h0[7]=(_Float16)f1.w;
    h1[0]=(_Float16)f2.x; h1[1]=(_Float16)f2.y; h1[2]=(_Float16)f2.z; h1[3]=(_Float16)f2.w;
    h1[4]=(_Float16)f3.x; h1[5]=(_Float16)f3.y; h1[6]=(_Float16)f3.z; h1[7]=(_Float16)f3.w;
    v8h* dst = (v8h*)(&As[0][fr * 32 + fseg]);
    dst[0] = h0; dst[1] = h1;
    wait_async();
  }
  __syncthreads();

  for (int kb = 0; kb < 32; ++kb) {
    const int cur = kb & 1, nxt = cur ^ 1;
    const bool have_next = (kb < 31);
    float4 f0, f1, f2, f3;
    if (have_next) {
      // issue next A-panel global loads (overlap with WMMAs below)
      const float4* s4 = (const float4*)(src + (size_t)(mbase + fr) * HIDDEN +
                                         (kb + 1) * 32 + fseg);
      f0 = s4[0]; f1 = s4[1]; f2 = s4[2]; f3 = s4[3];
      // issue next B-panel async-to-LDS
      const _Float16* g =
          wtw + (size_t)(nbase + fr) * HIDDEN + (kb + 1) * 32 + fseg;
      _Float16* l = &Bs[nxt][fr * 32 + fseg];
      async_copy_b128(g, l);
      async_copy_b128(g + 8, l + 8);
    }
    // ---- compute on current buffers
    v16h a[4], b[2];
#pragma unroll
    for (int mt = 0; mt < 4; mt++) {
      const _Float16* p = &As[cur][(waveM * 64 + mt * 16 + lh) * 32 + hig * 8];
      a[mt] = hcat(*(const v8h*)p, *(const v8h*)(p + 16));
    }
#pragma unroll
    for (int nt = 0; nt < 2; nt++) {
      const _Float16* p = &Bs[cur][(waveN * 32 + nt * 16 + lh) * 32 + hig * 8];
      b[nt] = hcat(*(const v8h*)p, *(const v8h*)(p + 16));
    }
#pragma unroll
    for (int mt = 0; mt < 4; mt++)
#pragma unroll
      for (int nt = 0; nt < 2; nt++)
        C[mt][nt] = wmma_f16(a[mt], b[nt], C[mt][nt]);

    if (have_next) {
      v8h h0, h1;
      h0[0]=(_Float16)f0.x; h0[1]=(_Float16)f0.y; h0[2]=(_Float16)f0.z; h0[3]=(_Float16)f0.w;
      h0[4]=(_Float16)f1.x; h0[5]=(_Float16)f1.y; h0[6]=(_Float16)f1.z; h0[7]=(_Float16)f1.w;
      h1[0]=(_Float16)f2.x; h1[1]=(_Float16)f2.y; h1[2]=(_Float16)f2.z; h1[3]=(_Float16)f2.w;
      h1[4]=(_Float16)f3.x; h1[5]=(_Float16)f3.y; h1[6]=(_Float16)f3.z; h1[7]=(_Float16)f3.w;
      v8h* dst = (v8h*)(&As[nxt][fr * 32 + fseg]);
      dst[0] = h0; dst[1] = h1;
      wait_async();
    }
    __syncthreads();
  }

  // ---- epilogue (uniform branch hoisted outside the unrolled loops)
  if (w == 2) {
#pragma unroll
    for (int mt = 0; mt < 4; mt++)
#pragma unroll
      for (int nt = 0; nt < 2; nt++)
#pragma unroll
        for (int v = 0; v < 8; v++) {
          int Mi = mbase + waveM * 64 + mt * 16 + v + hig * 8;
          int Ni = nbase + waveN * 32 + nt * 16 + lh;
          int bb = Mi >> 10, ss = Mi & 1023, hh = Ni >> 6, dd = Ni & 63;
          vtout[(((size_t)(bb * NHEADS + hh)) * DEPTH + dd) * SEQ + ss] =
              (_Float16)C[mt][nt][v];
        }
  } else {
    _Float16* outp = (w == 0) ? qout : kout;
    const float scale = (w == 0) ? 0.125f : 1.0f;  // DEPTH^-0.5 = 1/8
#pragma unroll
    for (int mt = 0; mt < 4; mt++)
#pragma unroll
      for (int nt = 0; nt < 2; nt++)
#pragma unroll
        for (int v = 0; v < 8; v++) {
          int Mi = mbase + waveM * 64 + mt * 16 + v + hig * 8;
          int Ni = nbase + waveN * 32 + nt * 16 + lh;
          int bb = Mi >> 10, ss = Mi & 1023, hh = Ni >> 6, dd = Ni & 63;
          outp[(((size_t)(bb * NHEADS + hh)) * SEQ + ss) * DEPTH + dd] =
              (_Float16)(C[mt][nt][v] * scale);
        }
  }
}

// ---------------------------------------------------------------------------
// Kernel 3: flash attention.  grid (S/64=16, B*H=64), block 128 (4 waves).
// Each wave: 16 q rows, streams key blocks of 32, online softmax, PV accum.
// Writes attnH fp16 [B*S][HIDDEN] (heads merged).
// ---------------------------------------------------------------------------
__global__ void __launch_bounds__(128) attn_kernel(
    const _Float16* __restrict__ Q, const _Float16* __restrict__ K,
    const _Float16* __restrict__ Vt, const float* __restrict__ bias,
    _Float16* __restrict__ attnH) {
  __shared__ _Float16 Pbuf[4][16 * 32];
  const int bh = blockIdx.y, bb = bh >> 4, hh = bh & 15;
  const int tid = threadIdx.x, wave = tid >> 5, lane = tid & 31;
  const int lh = lane & 15, hig = lane >> 4;
  const int qbase = blockIdx.x * 64 + wave * 16;
  const _Float16* Qb = Q + (size_t)bh * SEQ * DEPTH;
  const _Float16* Kb = K + (size_t)bh * SEQ * DEPTH;
  const _Float16* Vb = Vt + (size_t)bh * DEPTH * SEQ;
  _Float16* pb = Pbuf[wave];

  // preload Q fragments (A-layout), resident for whole kernel
  v16h aq[2];
#pragma unroll
  for (int kf = 0; kf < 2; kf++) {
    const _Float16* p = Qb + (size_t)(qbase + lh) * DEPTH + kf * 32 + hig * 8;
    aq[kf] = hcat(*(const v8h*)p, *(const v8h*)(p + 16));
  }

  v8f z = {0.f,0.f,0.f,0.f,0.f,0.f,0.f,0.f};
  v8f O[4] = {z, z, z, z};
  float mprev[8], lsum[8];
#pragma unroll
  for (int v = 0; v < 8; v++) { mprev[v] = -3.0e38f; lsum[v] = 0.f; }

  for (int kb = 0; kb < 32; kb++) {
    // ---- logits tile: 16q x 32keys, contraction over d=64 (2 WMMA / ntile)
    v8f Ct[2] = {z, z};
#pragma unroll
    for (int nt = 0; nt < 2; nt++) {
      const _Float16* p0 =
          Kb + (size_t)(kb * 32 + nt * 16 + lh) * DEPTH + hig * 8;
      v16h bk0 = hcat(*(const v8h*)p0, *(const v8h*)(p0 + 16));
      v16h bk1 = hcat(*(const v8h*)(p0 + 32), *(const v8h*)(p0 + 48));
      Ct[nt] = wmma_f16(aq[0], bk0, Ct[nt]);
      Ct[nt] = wmma_f16(aq[1], bk1, Ct[nt]);
      float bv = bias[bb * SEQ + kb * 32 + nt * 16 + lh];
#pragma unroll
      for (int v = 0; v < 8; v++) Ct[nt][v] += bv;
    }
    if (kb < 31) {  // global_prefetch_b8 for next K / V panels
      __builtin_prefetch(Kb + (size_t)((kb + 1) * 32 + lh) * DEPTH, 0, 1);
      __builtin_prefetch(Vb + (size_t)lh * SEQ + (kb + 1) * 32, 0, 1);
    }

    // ---- online softmax (rows on VGPR index v, cols across 16 lanes)
    float mb[8];
#pragma unroll
    for (int v = 0; v < 8; v++) mb[v] = fmaxf(Ct[0][v], Ct[1][v]);
#pragma unroll
    for (int mask = 1; mask < 16; mask <<= 1)
#pragma unroll
      for (int v = 0; v < 8; v++)
        mb[v] = fmaxf(mb[v], __shfl_xor(mb[v], mask, 32));
    float sc[8], rs[8];
#pragma unroll
    for (int v = 0; v < 8; v++) {
      float mn = fmaxf(mprev[v], mb[v]);
      sc[v] = __expf(mprev[v] - mn);
      mprev[v] = mn;
    }
#pragma unroll
    for (int v = 0; v < 8; v++) {
      float p0 = __expf(Ct[0][v] - mprev[v]);
      float p1 = __expf(Ct[1][v] - mprev[v]);
      Ct[0][v] = p0; Ct[1][v] = p1;
      rs[v] = p0 + p1;
    }
#pragma unroll
    for (int mask = 1; mask < 16; mask <<= 1)
#pragma unroll
      for (int v = 0; v < 8; v++) rs[v] += __shfl_xor(rs[v], mask, 32);
#pragma unroll
    for (int v = 0; v < 8; v++) lsum[v] = lsum[v] * sc[v] + rs[v];
#pragma unroll
    for (int dt = 0; dt < 4; dt++)
#pragma unroll
      for (int v = 0; v < 8; v++) O[dt][v] *= sc[v];

    // ---- C-layout -> A-layout via private LDS tile (per-wave, in-order LDS)
#pragma unroll
    for (int t = 0; t < 2; t++)
#pragma unroll
      for (int v = 0; v < 8; v++)
        pb[(v + hig * 8) * 32 + t * 16 + lh] = (_Float16)Ct[t][v];
    __builtin_amdgcn_wave_barrier();
    v16h pa;
    {
      const _Float16* p = pb + lh * 32 + hig * 8;
      pa = hcat(*(const v8h*)p, *(const v8h*)(p + 16));
    }
    __builtin_amdgcn_wave_barrier();

    // ---- O += P @ V   (B from Vt[d][s], contiguous in keys)
#pragma unroll
    for (int dt = 0; dt < 4; dt++) {
      const _Float16* p = Vb + (size_t)(dt * 16 + lh) * SEQ + kb * 32 + hig * 8;
      v16h bv = hcat(*(const v8h*)p, *(const v8h*)(p + 16));
      O[dt] = wmma_f16(pa, bv, O[dt]);
    }
  }

  // ---- normalize and write merged-head fp16 activations
#pragma unroll
  for (int v = 0; v < 8; v++) lsum[v] = 1.0f / lsum[v];
#pragma unroll
  for (int dt = 0; dt < 4; dt++)
#pragma unroll
    for (int v = 0; v < 8; v++) {
      int m = v + hig * 8;
      size_t row = (size_t)bb * SEQ + qbase + m;
      attnH[row * HIDDEN + hh * DEPTH + dt * 16 + lh] =
          (_Float16)(O[dt][v] * lsum[v]);
    }
}

// ---------------------------------------------------------------------------
// Kernel 4: output projection.  out = attnH(fp16) @ Wo -> fp32.
// grid (8, 32), block 256.  128x128 tile, both panels async double-buffered.
// ---------------------------------------------------------------------------
__global__ void __launch_bounds__(256) out_proj_kernel(
    const _Float16* __restrict__ attnH, const _Float16* __restrict__ wto,
    float* __restrict__ out) {
  __shared__ _Float16 As[2][128 * 32];
  __shared__ _Float16 Bs[2][128 * 32];
  const int mbase = blockIdx.y * 128, nbase = blockIdx.x * 128;
  const int tid = threadIdx.x, wave = tid >> 5, lane = tid & 31;
  const int lh = lane & 15, hig = lane >> 4;
  const int waveM = wave >> 2, waveN = wave & 3;
  const int fr = tid >> 1, fseg = (tid & 1) * 16;

  v8f z = {0.f,0.f,0.f,0.f,0.f,0.f,0.f,0.f};
  v8f C[4][2];
#pragma unroll
  for (int mt = 0; mt < 4; mt++) { C[mt][0] = z; C[mt][1] = z; }

  // prologue fill of buffer 0 (fully async)
  {
    const _Float16* ga = attnH + (size_t)(mbase + fr) * HIDDEN + fseg;
    _Float16* la = &As[0][fr * 32 + fseg];
    async_copy_b128(ga, la);
    async_copy_b128(ga + 8, la + 8);
    const _Float16* gb = wto + (size_t)(nbase + fr) * HIDDEN + fseg;
    _Float16* lb = &Bs[0][fr * 32 + fseg];
    async_copy_b128(gb, lb);
    async_copy_b128(gb + 8, lb + 8);
    wait_async();
  }
  __syncthreads();

  for (int kb = 0; kb < 32; ++kb) {
    const int cur = kb & 1, nxt = cur ^ 1;
    const bool have_next = (kb < 31);
    if (have_next) {
      const _Float16* ga =
          attnH + (size_t)(mbase + fr) * HIDDEN + (kb + 1) * 32 + fseg;
      _Float16* la = &As[nxt][fr * 32 + fseg];
      async_copy_b128(ga, la);
      async_copy_b128(ga + 8, la + 8);
      const _Float16* gb =
          wto + (size_t)(nbase + fr) * HIDDEN + (kb + 1) * 32 + fseg;
      _Float16* lb = &Bs[nxt][fr * 32 + fseg];
      async_copy_b128(gb, lb);
      async_copy_b128(gb + 8, lb + 8);
    }
    v16h a[4], b[2];
#pragma unroll
    for (int mt = 0; mt < 4; mt++) {
      const _Float16* p = &As[cur][(waveM * 64 + mt * 16 + lh) * 32 + hig * 8];
      a[mt] = hcat(*(const v8h*)p, *(const v8h*)(p + 16));
    }
#pragma unroll
    for (int nt = 0; nt < 2; nt++) {
      const _Float16* p = &Bs[cur][(waveN * 32 + nt * 16 + lh) * 32 + hig * 8];
      b[nt] = hcat(*(const v8h*)p, *(const v8h*)(p + 16));
    }
#pragma unroll
    for (int mt = 0; mt < 4; mt++)
#pragma unroll
      for (int nt = 0; nt < 2; nt++)
        C[mt][nt] = wmma_f16(a[mt], b[nt], C[mt][nt]);
    if (have_next) wait_async();
    __syncthreads();
  }
#pragma unroll
  for (int mt = 0; mt < 4; mt++)
#pragma unroll
    for (int nt = 0; nt < 2; nt++)
#pragma unroll
      for (int v = 0; v < 8; v++) {
        int Mi = mbase + waveM * 64 + mt * 16 + v + hig * 8;
        int Ni = nbase + waveN * 32 + nt * 16 + lh;
        out[(size_t)Mi * HIDDEN + Ni] = C[mt][nt][v];
      }
}

// ---------------------------------------------------------------------------
extern "C" void kernel_launch(void* const* d_in, const int* in_sizes, int n_in,
                              void* d_out, int out_size, void* d_ws,
                              size_t ws_size, hipStream_t stream) {
  const float* x    = (const float*)d_in[0];
  const float* y    = (const float*)d_in[1];
  const float* bias = (const float*)d_in[2];
  const float* Wq   = (const float*)d_in[3];
  const float* Wk   = (const float*)d_in[4];
  const float* Wv   = (const float*)d_in[5];
  const float* Wo   = (const float*)d_in[6];
  float* out = (float*)d_out;

  // workspace layout (halves): WT[4x1M] | Q | K | Vt | attnH   (total 40MB)
  _Float16* ws    = (_Float16*)d_ws;
  _Float16* wt    = ws;
  _Float16* Qb    = ws + (size_t)4  * 1024 * 1024;
  _Float16* Kbuf  = ws + (size_t)8  * 1024 * 1024;
  _Float16* Vtb   = ws + (size_t)12 * 1024 * 1024;
  _Float16* attnH = ws + (size_t)16 * 1024 * 1024;

  transpose_w_kernel<<<dim3(32, 32, 4), dim3(32, 8), 0, stream>>>(Wq, Wk, Wv,
                                                                  Wo, wt);
  qkv_proj_kernel<<<dim3(8, 32, 3), 256, 0, stream>>>(x, y, wt, Qb, Kbuf, Vtb);
  attn_kernel<<<dim3(16, 64), 128, 0, stream>>>(Qb, Kbuf, Vtb, bias, attnH);
  out_proj_kernel<<<dim3(8, 32), 256, 0, stream>>>(
      attnH, wt + (size_t)3 * 1024 * 1024, out);
}
